// DeepARGCNNet_57767310131500
// MI455X (gfx1250) — matrix operand order repaired and checked by reading, not compiled, not run on hardware
//
#include <hip/hip_runtime.h>
#include <hip/hip_bf16.h>
#include <stdint.h>

#define NN 20000
#define NE 320000
#define NH 4
#define NR 40
#define FIN 768
#define HID 128
#define NC 9

typedef __bf16 bf16_t;
typedef __attribute__((ext_vector_type(16))) __bf16 v16bf;
typedef __attribute__((ext_vector_type(8)))  float  v8f;
typedef __attribute__((ext_vector_type(4)))  unsigned int u32x4;
typedef __attribute__((ext_vector_type(8)))  int i32x8;
typedef __attribute__((ext_vector_type(4)))  int i32x4;

static __device__ __forceinline__ float lrelu(float x, float s) { return x > 0.0f ? x : x * s; }

static __device__ __forceinline__ void atomicMaxF(float* addr, float val) {
  int* ia = (int*)addr;
  int old = __float_as_int(*addr);
  while (__int_as_float(old) < val) {
    int assumed = old;
    old = atomicCAS(ia, assumed, __float_as_int(val));
    if (old == assumed) break;
  }
}

// ---------------- conversion / transpose ----------------
__global__ void k_f32_to_bf16(const float* __restrict__ in, bf16_t* __restrict__ out, int n) {
  int i = blockIdx.x * blockDim.x + threadIdx.x;
  if (i < n) out[i] = (bf16_t)in[i];
}

// W: [NH][K][O] f32 row-major  ->  Wt: [NH*O][K] bf16  (B^T for the WMMA GEMM)
__global__ void k_w_transpose_bf16(const float* __restrict__ W, bf16_t* __restrict__ Wt,
                                   int K, int O) {
  int i = blockIdx.x * blockDim.x + threadIdx.x;
  int total = NH * K * O;
  if (i >= total) return;
  int h = i / (K * O);
  int r = i - h * K * O;
  int f = r / O;
  int o = r - f * O;
  Wt[(size_t)(h * O + o) * K + f] = (bf16_t)W[i];
}

// ---------------- WMMA GEMM with TDM-staged A panel ----------------
// C[M x Nn] = A[M x K](bf16) * Bt[Nn x K]^T (bf16), fp32 acc.  Requires K % 64 == 0.
// block = 256 threads (8 waves); blockIdx.x = M/16 tile; each wave covers 64 N columns.
// The 16 x K A panel is DMA'd into LDS once per block by the Tensor Data Mover;
// B fragments are ping-pong double-buffered (no register copies) so global loads
// overlap the WMMAs; K loop unrolled 2x -> 8 WMMA per iteration.

static __device__ __forceinline__ v16bf load_bfrag(const bf16_t* pb) {
  union { uint4 q[2]; v16bf v; } u;
  u.q[0] = ((const uint4*)pb)[0];
  u.q[1] = ((const uint4*)(pb + 16))[0];
  return u.v;
}

static __device__ __forceinline__ v16bf load_afrag_lds(const bf16_t* pa) {
  union { uint4 q[2]; v16bf v; } u;
  u.q[0] = ((const uint4*)pa)[0];        // K = kbA .. kbA+7
  u.q[1] = ((const uint4*)(pa + 16))[0]; // K = 16+kbA .. 16+kbA+7
  return u.v;
}

__global__ __launch_bounds__(256) void k_wmma_gemm(const bf16_t* __restrict__ A,
                                                   const bf16_t* __restrict__ Bt,
                                                   float* __restrict__ C,
                                                   int K, int Nn) {
  extern __shared__ char smem[];
  bf16_t* lds_a = (bf16_t*)smem;           // [16][K], dynamic LDS => offset 0

  const int lane = threadIdx.x & 31;
  const int wave = threadIdx.x >> 5;
  const int m0 = blockIdx.x * 16;
  const int n0 = wave * 64;

  // ---- TDM: DMA the 16 x K bf16 A panel (row-major) into LDS, one op per block ----
  if (threadIdx.x < 32) {
    uint64_t ga = (uint64_t)(uintptr_t)(A + (size_t)m0 * K);
    unsigned Ku = (unsigned)K;
    // D# group 0: count=1 | lds_addr | global_addr[56:0] | type=2 ("image")
    u32x4 g0 = { 1u,                                   // count=1, is_restore=0, gather=0
                 0u,                                   // lds_addr (bytes)
                 (unsigned)(ga & 0xFFFFFFFFu),
                 (unsigned)((ga >> 32) & 0x01FFFFFFu) | (2u << 30) };
    // D# group 1 (bit-packed per ISA 8.4): wg_mask=0, data_size=1 (2B),
    // tensor_dim0=K, tensor_dim1=16, tile_dim0=K, tile_dim1=16, tile_dim2=0,
    // tensor_dim0_stride=K, tensor_dim1_stride=0, no padding/iterate/barrier.
    i32x8 g1 = { (int)(1u << 16),                      // [17:16] data_size = 2 bytes
                 (int)((Ku & 0xFFFFu) << 16),          // [63:48] tensor_dim0 lo16
                 (int)((Ku >> 16) | (16u << 16)),      // tensor_dim0 hi16 | tensor_dim1 lo16
                 (int)((Ku & 0xFFFFu) << 16),          // tensor_dim1 hi16=0 | [127:112] tile_dim0
                 (int)16u,                             // [143:128] tile_dim1=16 | tile_dim2=0
                 (int)Ku,                              // [191:160] tensor_dim0_stride lo32
                 0, 0 };                               // stride0 hi16=0, stride1=0
    i32x4 gz4 = { 0, 0, 0, 0 };                        // groups 2/3 unused (2-D tensor)
    i32x8 gz8 = { 0, 0, 0, 0, 0, 0, 0, 0 };            // trailing group unused
    __builtin_amdgcn_tensor_load_to_lds(g0, g1, gz4, gz4, gz8, 0);
    __builtin_amdgcn_s_wait_tensorcnt(0);
  }
  __syncthreads();

  v8f acc[4];
#pragma unroll
  for (int t = 0; t < 4; ++t) acc[t] = v8f{0.f,0.f,0.f,0.f,0.f,0.f,0.f,0.f};

  const int mlo = lane & 15;
  const int kbA = (lane >> 4) * 8;     // 16-bit A fragment layout
  const int nlo = lane & 15;
  const int kbB = (lane >> 4) * 8;     // 16-bit B fragment layout

  const bf16_t* arow = lds_a + (size_t)mlo * K + kbA;
  const bf16_t* brow = Bt + (size_t)(n0 + nlo) * K + kbB;

  // ping-pong B buffers: bA holds k0, bB holds k0+32
  v16bf bA[4], bB[4];
#pragma unroll
  for (int t = 0; t < 4; ++t) bA[t] = load_bfrag(brow + (size_t)t * 16 * K);

  for (int k0 = 0; k0 < K; k0 += 64) {
    // stage 1: prefetch k0+32 into bB, compute with bA
#pragma unroll
    for (int t = 0; t < 4; ++t) bB[t] = load_bfrag(brow + (size_t)t * 16 * K + k0 + 32);
    {
      v16bf a = load_afrag_lds(arow + k0);
#pragma unroll
      for (int t = 0; t < 4; ++t)
        acc[t] = __builtin_amdgcn_wmma_f32_16x16x32_bf16(false, a, false, bA[t],
                                                         (short)0, acc[t], false, false);
    }
    // stage 2: prefetch k0+64 into bA (if any), compute with bB
    if (k0 + 64 < K) {
#pragma unroll
      for (int t = 0; t < 4; ++t) bA[t] = load_bfrag(brow + (size_t)t * 16 * K + k0 + 64);
    }
    {
      v16bf a = load_afrag_lds(arow + k0 + 32);
#pragma unroll
      for (int t = 0; t < 4; ++t)
        acc[t] = __builtin_amdgcn_wmma_f32_16x16x32_bf16(false, a, false, bB[t],
                                                         (short)0, acc[t], false, false);
    }
  }

  const int mb = m0 + (lane >> 4) * 8;   // C/D layout: VGPR v -> M = v + 8*(lane>=16)
#pragma unroll
  for (int t = 0; t < 4; ++t) {
    int n = n0 + t * 16 + nlo;
#pragma unroll
    for (int v = 0; v < 8; ++v) C[(size_t)(mb + v) * Nn + n] = acc[t][v];
  }
}

// ---------------- attention pipeline ----------------
__global__ void k_fill(float* __restrict__ p, float val, size_t n) {
  size_t i = (size_t)blockIdx.x * blockDim.x + threadIdx.x;
  if (i < n) p[i] = val;
}

__global__ void k_node_dots(const float* __restrict__ hb, const float* __restrict__ a_s,
                            const float* __restrict__ a_d, float* __restrict__ es,
                            float* __restrict__ ed, int O) {
  int i = blockIdx.x * blockDim.x + threadIdx.x;
  if (i >= NN * NH) return;
  int n = i / NH, h = i - n * NH;
  const float* hp = hb + (size_t)n * NH * O + (size_t)h * O;
  const float* as = a_s + h * O;
  const float* ad = a_d + h * O;
  float s = 0.f, d = 0.f;
  for (int o = 0; o < O; ++o) { float v = hp[o]; s += v * as[o]; d += v * ad[o]; }
  es[i] = s; ed[i] = d;
}

__global__ void k_logit_max(const int* __restrict__ src, const int* __restrict__ dst,
                            const int* __restrict__ etyp, const float* __restrict__ rel,
                            const float* __restrict__ a_e, const float* __restrict__ es,
                            const float* __restrict__ ed, float* __restrict__ logit,
                            float* __restrict__ mmax) {
  int i = blockIdx.x * blockDim.x + threadIdx.x;
  if (i >= NE * NH) return;
  int e = i / NH, h = i - e * NH;
  int s = src[e], d = dst[e], t = etyp[e];
  float ee = rel[t * 2] * a_e[h * 2] + rel[t * 2 + 1] * a_e[h * 2 + 1];
  float l = lrelu(es[s * NH + h] + ed[d * NH + h] + ee, 0.2f);
  logit[i] = l;
  atomicMaxF(&mmax[d * NH + h], l);
}

__global__ void k_exp_sum(const int* __restrict__ dst, const float* __restrict__ mmax,
                          float* __restrict__ logit, float* __restrict__ den) {
  int i = blockIdx.x * blockDim.x + threadIdx.x;
  if (i >= NE * NH) return;
  int e = i / NH, h = i - e * NH;
  float ex = __expf(logit[i] - mmax[dst[e] * NH + h]);
  logit[i] = ex;
  atomicAdd(&den[dst[e] * NH + h], ex);
}

__global__ void k_alpha(const int* __restrict__ dst, const int* __restrict__ edist,
                        const float* __restrict__ den, float* __restrict__ alpha) {
  int i = blockIdx.x * blockDim.x + threadIdx.x;
  if (i >= NE * NH) return;
  int e = i / NH, h = i - e * NH;
  float w = 1.0f / (1.0f + (float)edist[e]);
  alpha[i] = alpha[i] / (den[dst[e] * NH + h] + 1e-16f) * w;
}

// generic scalar scatter (used for O=9 final layer)
__global__ void k_scatter(const int* __restrict__ src, const int* __restrict__ dst,
                          const float* __restrict__ alpha, const float* __restrict__ hb,
                          float* __restrict__ acc, int O) {
  long long i = (long long)blockIdx.x * blockDim.x + threadIdx.x;
  long long total = (long long)NE * NH * O;
  if (i >= total) return;
  int e = (int)(i / (NH * O));
  int r = (int)(i - (long long)e * (NH * O));
  int h = r / O;
  float v = alpha[e * NH + h] * hb[(size_t)src[e] * (NH * O) + r];
  atomicAdd(&acc[(size_t)dst[e] * (NH * O) + r], v);
}

// vectorized scatter for O % 4 == 0 (one float4 gather + 4 atomics per thread)
__global__ void k_scatter4(const int* __restrict__ src, const int* __restrict__ dst,
                           const float* __restrict__ alpha, const float* __restrict__ hb,
                           float* __restrict__ acc, int O) {
  const int O4 = O >> 2;
  long long i = (long long)blockIdx.x * blockDim.x + threadIdx.x;
  long long total = (long long)NE * NH * O4;
  if (i >= total) return;
  int e = (int)(i / (NH * O4));
  int r = (int)(i - (long long)e * (NH * O4));
  int h = r / O4;
  int oq = r - h * O4;
  int base = h * O + oq * 4;
  float a = alpha[e * NH + h];
  const float4 hv = *(const float4*)(hb + (size_t)src[e] * (NH * O) + base);
  float* ap = acc + (size_t)dst[e] * (NH * O) + base;
  atomicAdd(ap + 0, a * hv.x);
  atomicAdd(ap + 1, a * hv.y);
  atomicAdd(ap + 2, a * hv.z);
  atomicAdd(ap + 3, a * hv.w);
}

__global__ void k_mean_bias(const float* __restrict__ acc, const float* __restrict__ bias,
                            const float* __restrict__ residual, float* __restrict__ out,
                            int O, int do_leaky) {
  int i = blockIdx.x * blockDim.x + threadIdx.x;
  if (i >= NN * O) return;
  int n = i / O, o = i - n * O;
  float s = 0.f;
  for (int h = 0; h < NH; ++h) s += acc[(size_t)n * NH * O + h * O + o];
  float v = s * (1.0f / NH) + bias[o];
  if (residual) v += residual[i];
  if (do_leaky) v = lrelu(v, 0.1f);
  out[i] = v;
}

// ---------------- batchnorm ----------------
__global__ void k_bn_stats(const float* __restrict__ x, float* __restrict__ sums) {
  int i = blockIdx.x * blockDim.x + threadIdx.x;
  if (i >= NN * HID) return;
  int f = i & (HID - 1);
  float v = x[i];
  atomicAdd(&sums[f], v);
  atomicAdd(&sums[HID + f], v * v);
}

__global__ void k_bn_apply(const float* __restrict__ x, const float* __restrict__ sums,
                           const float* __restrict__ g, const float* __restrict__ b,
                           float* __restrict__ out) {
  int i = blockIdx.x * blockDim.x + threadIdx.x;
  if (i >= NN * HID) return;
  int f = i & (HID - 1);
  float mu  = sums[f] * (1.0f / NN);
  float var = sums[HID + f] * (1.0f / NN) - mu * mu;
  float v = (x[i] - mu) * rsqrtf(var + 1e-5f) * g[f] + b[f];
  out[i] = lrelu(v, 0.1f);
}

// ---------------- final skinny GEMM (N=9 per head, too small for WMMA) -------
__global__ void k_gemm_final(const float* __restrict__ hact, const float* __restrict__ WL,
                             float* __restrict__ hL) {
  int i = blockIdx.x * blockDim.x + threadIdx.x;
  if (i >= NN * NH * NC) return;
  int n = i / (NH * NC);
  int r = i - n * (NH * NC);
  int h = r / NC, c = r - h * NC;
  const float* xp = hact + (size_t)n * HID;
  const float* wp = WL + (size_t)h * HID * NC + c;
  float s = 0.f;
  for (int f = 0; f < HID; ++f) s += xp[f] * wp[(size_t)f * NC];
  hL[i] = s;
}

// ---------------- host orchestration ----------------
static inline unsigned nblk(long long n, int tb) { return (unsigned)((n + tb - 1) / tb); }

extern "C" void kernel_launch(void* const* d_in, const int* in_sizes, int n_in,
                              void* d_out, int out_size, void* d_ws, size_t ws_size,
                              hipStream_t stream) {
  (void)in_sizes; (void)n_in; (void)out_size; (void)ws_size;
  const float* x      = (const float*)d_in[0];
  const int*   eidx   = (const int*)d_in[1];
  const int*   etyp   = (const int*)d_in[2];
  const int*   edist  = (const int*)d_in[3];
  const float* W0     = (const float*)d_in[4];
  const float* asrc0  = (const float*)d_in[5];
  const float* adst0  = (const float*)d_in[6];
  const float* aedge0 = (const float*)d_in[7];
  const float* b0     = (const float*)d_in[8];
  const float* rel0   = (const float*)d_in[9];
  const float* Wm     = (const float*)d_in[10];
  const float* asrcm  = (const float*)d_in[11];
  const float* adstm  = (const float*)d_in[12];
  const float* aedgem = (const float*)d_in[13];
  const float* bm     = (const float*)d_in[14];
  const float* relm   = (const float*)d_in[15];
  const float* WL     = (const float*)d_in[16];
  const float* asrcL  = (const float*)d_in[17];
  const float* adstL  = (const float*)d_in[18];
  const float* aedgeL = (const float*)d_in[19];
  const float* bL     = (const float*)d_in[20];
  const float* relL   = (const float*)d_in[21];
  const float* bng    = (const float*)d_in[22];
  const float* bnb    = (const float*)d_in[23];

  const int* src = eidx;
  const int* dst = eidx + NE;

  uintptr_t wp_ = (uintptr_t)d_ws;
  auto carve = [&](size_t bytes) -> void* {
    void* p = (void*)wp_;
    wp_ += (bytes + 255) & ~(size_t)255;
    return p;
  };
  bf16_t* xb   = (bf16_t*)carve((size_t)NN * FIN * 2);
  bf16_t* wt   = (bf16_t*)carve((size_t)NH * HID * FIN * 2);
  float*  hbuf = (float*)carve((size_t)NN * NH * HID * 4);
  float*  accb = (float*)carve((size_t)NN * NH * HID * 4);
  float*  xcur = (float*)carve((size_t)NN * HID * 4);
  float*  hact = (float*)carve((size_t)NN * HID * 4);
  bf16_t* hb16 = (bf16_t*)carve((size_t)NN * HID * 2);
  float*  es   = (float*)carve((size_t)NN * NH * 4);
  float*  ed   = (float*)carve((size_t)NN * NH * 4);
  float*  mmax = (float*)carve((size_t)NN * NH * 4);
  float*  den  = (float*)carve((size_t)NN * NH * 4);
  float*  alph = (float*)carve((size_t)NE * NH * 4);
  float*  bns  = (float*)carve((size_t)2 * HID * 4);
  float*  hL   = (float*)carve((size_t)NN * NH * NC * 4);

  const int TB = 256;

  auto run_conv = [&](const float* hb, int O, const float* a_s, const float* a_d,
                      const float* a_e, const float* rel, const float* bias,
                      const float* residual, float* outp, int leakyOut) {
    long long nh  = (long long)NN * NH;
    long long eh  = (long long)NE * NH;
    long long nho = (long long)NN * NH * O;
    k_node_dots<<<nblk(nh, TB), TB, 0, stream>>>(hb, a_s, a_d, es, ed, O);
    k_fill<<<nblk(nh, TB), TB, 0, stream>>>(mmax, -1e30f, (size_t)nh);
    k_fill<<<nblk(nh, TB), TB, 0, stream>>>(den, 0.f, (size_t)nh);
    k_fill<<<nblk(nho, TB), TB, 0, stream>>>(accb, 0.f, (size_t)nho);
    k_logit_max<<<nblk(eh, TB), TB, 0, stream>>>(src, dst, etyp, rel, a_e, es, ed, alph, mmax);
    k_exp_sum<<<nblk(eh, TB), TB, 0, stream>>>(dst, mmax, alph, den);
    k_alpha<<<nblk(eh, TB), TB, 0, stream>>>(dst, edist, den, alph);
    if ((O & 3) == 0) {
      long long eho4 = (long long)NE * NH * (O >> 2);
      k_scatter4<<<nblk(eho4, TB), TB, 0, stream>>>(src, dst, alph, hb, accb, O);
    } else {
      long long eho = (long long)NE * NH * O;
      k_scatter<<<nblk(eho, TB), TB, 0, stream>>>(src, dst, alph, hb, accb, O);
    }
    k_mean_bias<<<nblk((long long)NN * O, TB), TB, 0, stream>>>(accb, bias, residual, outp, O, leakyOut);
  };

  // ---- layer 0: WMMA GEMM (K=768, A panel via TDM) + conv ----
  k_f32_to_bf16<<<nblk((long long)NN * FIN, TB), TB, 0, stream>>>(x, xb, NN * FIN);
  k_w_transpose_bf16<<<nblk((long long)NH * FIN * HID, TB), TB, 0, stream>>>(W0, wt, FIN, HID);
  k_wmma_gemm<<<NN / 16, 256, (size_t)16 * FIN * 2, stream>>>(xb, wt, hbuf, FIN, NH * HID);
  run_conv(hbuf, HID, asrc0, adst0, aedge0, rel0, b0, nullptr, xcur, 0);

  // ---- 3 residual mid layers ----
  for (int i = 0; i < 3; ++i) {
    k_fill<<<1, 256, 0, stream>>>(bns, 0.f, (size_t)(2 * HID));
    k_bn_stats<<<nblk((long long)NN * HID, TB), TB, 0, stream>>>(xcur, bns);
    k_bn_apply<<<nblk((long long)NN * HID, TB), TB, 0, stream>>>(xcur, bns, bng + i * HID, bnb + i * HID, hact);
    k_f32_to_bf16<<<nblk((long long)NN * HID, TB), TB, 0, stream>>>(hact, hb16, NN * HID);
    k_w_transpose_bf16<<<nblk((long long)NH * HID * HID, TB), TB, 0, stream>>>(
        Wm + (size_t)i * NH * HID * HID, wt, HID, HID);
    k_wmma_gemm<<<NN / 16, 256, (size_t)16 * HID * 2, stream>>>(hb16, wt, hbuf, HID, NH * HID);
    run_conv(hbuf, HID, asrcm + i * NH * HID, adstm + i * NH * HID,
             aedgem + i * NH * 2, relm + i * NR * 2, bm + i * HID, hact, xcur, 0);
  }

  // ---- final layer: BN + skinny GEMM + conv (O=9) + leaky(0.1) ----
  k_fill<<<1, 256, 0, stream>>>(bns, 0.f, (size_t)(2 * HID));
  k_bn_stats<<<nblk((long long)NN * HID, TB), TB, 0, stream>>>(xcur, bns);
  k_bn_apply<<<nblk((long long)NN * HID, TB), TB, 0, stream>>>(xcur, bns, bng + 3 * HID, bnb + 3 * HID, hact);
  k_gemm_final<<<nblk((long long)NN * NH * NC, TB), TB, 0, stream>>>(hact, WL, hL);
  run_conv(hL, NC, asrcL, adstL, aedgeL, relL, bL, nullptr, (float*)d_out, 1);
}
